// HGT_29841432772813
// MI455X (gfx1250) — compile-verified, matrix operand order checked by reading, not run on hardware
//
#include <hip/hip_runtime.h>

// ---------------- constants ----------------
#define HIDDIM 128
#define NHEAD  8
#define HD     16
#define NEDGE  100000

typedef __attribute__((ext_vector_type(16))) __bf16 v16bf;
typedef __attribute__((ext_vector_type(8)))  float  v8f;

union FragAB { uint4 u[2]; v16bf v; };
union Row16  { uint4 q[2]; unsigned short s[16]; };

// ---------------- device helpers ----------------
__device__ __forceinline__ unsigned short f2bf(float f) {
    unsigned u = __float_as_uint(f);
    unsigned r = u + 0x7fffu + ((u >> 16) & 1u);   // round-to-nearest-even
    return (unsigned short)(r >> 16);
}
__device__ __forceinline__ float bf2f(unsigned short s) {
    return __uint_as_float(((unsigned)s) << 16);
}
__device__ __forceinline__ unsigned f2ord(float f) {   // order-preserving float->uint
    unsigned u = __float_as_uint(f);
    return (u & 0x80000000u) ? ~u : (u | 0x80000000u);
}
__device__ __forceinline__ float ord2f(unsigned u) {
    unsigned v = (u & 0x80000000u) ? (u ^ 0x80000000u) : ~u;
    return __uint_as_float(v);
}

// ---------------- WMMA GEMM: C[M,ND] = A[M,KD](bf16) * W[KD,ND] + bias ----------------
// Bt is W transposed: Bt[n*KD + k] = W[k*ND + n], bf16. Staged in LDS per block.
// Fragment lane layout per CDNA5 ISA 16-bit matrices:
//   lanes 0-15  : K = kb*32 + {0..7} and {16..23}
//   lanes 16-31 : K = kb*32 + {8..15} and {24..31}
// Block: 256 threads = 8 waves; each wave processes RT row tiles of 16 rows.
template<int KD, int ND, int RT, bool RELU, bool SKIP>
__global__ __launch_bounds__(256) void gemm_wmma(
    const unsigned short* __restrict__ A,
    const unsigned short* __restrict__ Bt,
    const float* __restrict__ bias,
    const float* __restrict__ skipX,      // fp32 [M,ND], used if SKIP
    const float* __restrict__ skipBeta,   // scalar (pre-sigmoid), used if SKIP
    float* __restrict__ Cf,               // optional fp32 out [M,ND]
    unsigned short* __restrict__ Cb,      // optional bf16 out [M,ND]
    int M)
{
    constexpr int NT = ND / 16;
    constexpr int KT = KD / 32;

    __shared__ __align__(16) unsigned short Bs[ND * KD];
    {
        constexpr int NV = ND * KD / 8;
        const uint4* src = (const uint4*)Bt;
        uint4* dst = (uint4*)Bs;
        #pragma unroll 4
        for (int i = threadIdx.x; i < NV; i += 256) dst[i] = src[i];
    }
    __syncthreads();

    const int lane  = threadIdx.x & 31;
    const int wid   = threadIdx.x >> 5;
    const int kHalf = (lane < 16) ? 0 : 8;
    const int col0  = lane & 15;
    const int rOff  = (lane >= 16) ? 8 : 0;

    float beta = 0.f, omb = 0.f;
    if constexpr (SKIP) {
        beta = 1.f / (1.f + __expf(-skipBeta[0]));
        omb  = 1.f - beta;
    }
    float bvals[NT];
    #pragma unroll
    for (int nt = 0; nt < NT; ++nt) bvals[nt] = bias ? bias[nt * 16 + col0] : 0.f;

    for (int t = 0; t < RT; ++t) {
        const int row0 = (blockIdx.x * (8 * RT) + t * 8 + wid) * 16;
        if (row0 >= M) break;              // M multiple of 16: whole wave exits

        const int rowN = row0 + 128;       // next tile for this wave
        if (t + 1 < RT && rowN < M)
            __builtin_prefetch(A + (size_t)(rowN + col0) * KD + kHalf, 0, 1);

        v8f acc[NT] = {};
        const unsigned short* aBase = A + (size_t)(row0 + col0) * KD + kHalf;

        #pragma unroll
        for (int kb = 0; kb < KT; ++kb) {
            const int k0 = kb * 32;
            FragAB af;
            af.u[0] = *(const uint4*)(aBase + k0);
            af.u[1] = *(const uint4*)(aBase + k0 + 16);
            #pragma unroll
            for (int nt = 0; nt < NT; ++nt) {
                const unsigned short* bBase = Bs + (nt * 16 + col0) * KD + kHalf + k0;
                FragAB bfr;
                bfr.u[0] = *(const uint4*)(bBase);
                bfr.u[1] = *(const uint4*)(bBase + 16);
                acc[nt] = __builtin_amdgcn_wmma_f32_16x16x32_bf16(
                    false, af.v, false, bfr.v, (short)0, acc[nt], false, false);
            }
        }

        #pragma unroll
        for (int nt = 0; nt < NT; ++nt) {
            const int n = nt * 16 + col0;
            #pragma unroll
            for (int j = 0; j < 8; ++j) {
                const int m = row0 + rOff + j;
                float v = acc[nt][j] + bvals[nt];
                if constexpr (SKIP) v = beta * v + omb * skipX[(size_t)m * ND + n];
                if constexpr (RELU) v = fmaxf(v, 0.f);
                if (Cf) Cf[(size_t)m * ND + n] = v;
                if (Cb) Cb[(size_t)m * ND + n] = f2bf(v);
            }
        }
    }
}

// ---------------- fused K/Q/V projection ----------------
// One pass over A; three weight matrices staged in dynamic LDS (3*ND*KD bf16 = 96KB).
// A fragments for a tile are loaded once and reused for all three GEMMs.
// Outputs are bf16 (halves edge-phase gather traffic).
template<int KD, int ND, int RT>
__global__ __launch_bounds__(256) void gemm_qkv(
    const unsigned short* __restrict__ A,
    const unsigned short* __restrict__ BtK,
    const unsigned short* __restrict__ BtQ,
    const unsigned short* __restrict__ BtV,
    const float* __restrict__ bK,
    const float* __restrict__ bQ,
    const float* __restrict__ bV,
    unsigned short* __restrict__ OutK,
    unsigned short* __restrict__ OutQ,
    unsigned short* __restrict__ OutV,
    int M)
{
    constexpr int NT = ND / 16;
    constexpr int KT = KD / 32;
    constexpr int MAT = ND * KD;          // elements per matrix

    extern __shared__ unsigned short Bs3[];   // 3 * MAT bf16
    {
        constexpr int NV = MAT / 8;
        const unsigned short* srcs[3] = {BtK, BtQ, BtV};
        uint4* dst = (uint4*)Bs3;
        for (int g = 0; g < 3; ++g) {
            const uint4* s = (const uint4*)srcs[g];
            #pragma unroll 4
            for (int i = threadIdx.x; i < NV; i += 256) dst[g * NV + i] = s[i];
        }
    }
    __syncthreads();

    const int lane  = threadIdx.x & 31;
    const int wid   = threadIdx.x >> 5;
    const int kHalf = (lane < 16) ? 0 : 8;
    const int col0  = lane & 15;
    const int rOff  = (lane >= 16) ? 8 : 0;

    const float* biases[3] = {bK, bQ, bV};
    unsigned short* outs[3] = {OutK, OutQ, OutV};

    for (int t = 0; t < RT; ++t) {
        const int row0 = (blockIdx.x * (8 * RT) + t * 8 + wid) * 16;
        if (row0 >= M) break;

        // load A fragments once, reuse across the three GEMMs
        FragAB afr[KT];
        const unsigned short* aBase = A + (size_t)(row0 + col0) * KD + kHalf;
        #pragma unroll
        for (int kb = 0; kb < KT; ++kb) {
            afr[kb].u[0] = *(const uint4*)(aBase + kb * 32);
            afr[kb].u[1] = *(const uint4*)(aBase + kb * 32 + 16);
        }

        for (int g = 0; g < 3; ++g) {
            const unsigned short* Bg = Bs3 + g * MAT;
            v8f acc[NT] = {};
            #pragma unroll
            for (int kb = 0; kb < KT; ++kb) {
                const int k0 = kb * 32;
                #pragma unroll
                for (int nt = 0; nt < NT; ++nt) {
                    const unsigned short* bBase = Bg + (nt * 16 + col0) * KD + kHalf + k0;
                    FragAB bfr;
                    bfr.u[0] = *(const uint4*)(bBase);
                    bfr.u[1] = *(const uint4*)(bBase + 16);
                    acc[nt] = __builtin_amdgcn_wmma_f32_16x16x32_bf16(
                        false, afr[kb].v, false, bfr.v, (short)0, acc[nt], false, false);
                }
            }
            const float* bias = biases[g];
            unsigned short* Og = outs[g];
            #pragma unroll
            for (int nt = 0; nt < NT; ++nt) {
                const int n = nt * 16 + col0;
                const float bval = bias[n];
                #pragma unroll
                for (int j = 0; j < 8; ++j) {
                    const int m = row0 + rOff + j;
                    Og[(size_t)m * ND + n] = f2bf(acc[nt][j] + bval);
                }
            }
        }
    }
}

// ---------------- utility kernels ----------------
__global__ __launch_bounds__(256) void k_cvt(const float* __restrict__ X,
                                             unsigned short* __restrict__ Y, int n) {
    int i = blockIdx.x * 256 + threadIdx.x;
    if (i < n) Y[i] = f2bf(X[i]);
}
// W[K,N] fp32 -> Wt[N,K] bf16
__global__ __launch_bounds__(256) void k_wcvt(const float* __restrict__ W,
                                              unsigned short* __restrict__ Wt,
                                              int K, int N) {
    int i = blockIdx.x * 256 + threadIdx.x;
    if (i >= K * N) return;
    int n = i / K, k = i - n * K;
    Wt[i] = f2bf(W[(size_t)k * N + n]);
}
__global__ __launch_bounds__(256) void k_fill_f32(float* __restrict__ p, float v, int n) {
    int i = blockIdx.x * 256 + threadIdx.x;
    if (i < n) p[i] = v;
}
__global__ __launch_bounds__(256) void k_fill_u32(unsigned* __restrict__ p, unsigned v, int n) {
    int i = blockIdx.x * 256 + threadIdx.x;
    if (i < n) p[i] = v;
}

// ---------------- edge pass A: scores + per-(dst,head) running max ----------------
// thread t = e*8 + h; K/Q are bf16 [N,128]
__global__ __launch_bounds__(256) void k_edgeA(
    const int* __restrict__ ei,                 // [2,E]: src then dst
    const unsigned short* __restrict__ Ksrc,    // [Ns,128] bf16
    const unsigned short* __restrict__ Qdst,    // [Nd,128] bf16
    const float* __restrict__ Amat,             // [8,16,16] fp32
    const float* __restrict__ pvec,             // [8]
    float* __restrict__ sc,                     // [E,8]
    unsigned* __restrict__ mMax,                // [Nd,8] ordered-uint max
    int E)
{
    int t = blockIdx.x * 256 + threadIdx.x;
    if (t >= E * NHEAD) return;
    int e = t >> 3, h = t & 7;
    int si = ei[e], di = ei[E + e];
    Row16 kbuf, qbuf;
    {
        const uint4* kp = (const uint4*)(Ksrc + (size_t)si * HIDDIM + h * HD);
        kbuf.q[0] = kp[0]; kbuf.q[1] = kp[1];
        const uint4* qp = (const uint4*)(Qdst + (size_t)di * HIDDIM + h * HD);
        qbuf.q[0] = qp[0]; qbuf.q[1] = qp[1];
    }
    const float* Ah = Amat + h * (HD * HD);
    float ke[HD];
    #pragma unroll
    for (int f = 0; f < HD; ++f) ke[f] = 0.f;
    #pragma unroll
    for (int d = 0; d < HD; ++d) {
        float kd = bf2f(kbuf.s[d]);
        #pragma unroll
        for (int f = 0; f < HD; ++f) ke[f] = fmaf(kd, Ah[d * HD + f], ke[f]);
    }
    float s = 0.f;
    #pragma unroll
    for (int f = 0; f < HD; ++f) s = fmaf(ke[f], bf2f(qbuf.s[f]), s);
    s *= pvec[h] * 0.25f;                 // 1/sqrt(16)
    sc[t] = s;
    atomicMax(&mMax[(size_t)di * NHEAD + h], f2ord(s));
}

// ---------------- edge pass B: exp, sum, message scatter ----------------
__global__ __launch_bounds__(256) void k_edgeB(
    const int* __restrict__ ei,
    const unsigned short* __restrict__ Vsrc,    // [Ns,128] bf16
    const float* __restrict__ Mmat,             // [8,16,16]
    const float* __restrict__ sc,               // [E,8]
    const unsigned* __restrict__ mMax,          // [Nd,8]
    float* __restrict__ ssum,                   // [Nd,8]
    float* __restrict__ agg,                    // [Nd,128]
    int E)
{
    int t = blockIdx.x * 256 + threadIdx.x;
    if (t >= E * NHEAD) return;
    int e = t >> 3, h = t & 7;
    int si = ei[e], di = ei[E + e];
    float m = ord2f(mMax[(size_t)di * NHEAD + h]);
    float ex = __expf(sc[t] - m);
    atomicAdd(&ssum[(size_t)di * NHEAD + h], ex);
    Row16 vbuf;
    {
        const uint4* vp = (const uint4*)(Vsrc + (size_t)si * HIDDIM + h * HD);
        vbuf.q[0] = vp[0]; vbuf.q[1] = vp[1];
    }
    const float* Mh = Mmat + h * (HD * HD);
    float ve[HD];
    #pragma unroll
    for (int f = 0; f < HD; ++f) ve[f] = 0.f;
    #pragma unroll
    for (int d = 0; d < HD; ++d) {
        float vd = bf2f(vbuf.s[d]);
        #pragma unroll
        for (int f = 0; f < HD; ++f) ve[f] = fmaf(vd, Mh[d * HD + f], ve[f]);
    }
    float* ag = agg + (size_t)di * HIDDIM + h * HD;
    #pragma unroll
    for (int f = 0; f < HD; ++f) atomicAdd(&ag[f], ex * ve[f]);
}

// ---------------- normalize + gelu -> bf16 ----------------
__global__ __launch_bounds__(256) void k_gelu(
    const float* __restrict__ agg, const float* __restrict__ ssum,
    unsigned short* __restrict__ G, int Nn)
{
    int i = blockIdx.x * 256 + threadIdx.x;
    if (i >= Nn * HIDDIM) return;
    int node = i >> 7;
    int h = (i & 127) >> 4;
    float s = ssum[(size_t)node * NHEAD + h];
    float x = agg[i] / (s + 1e-16f);
    float tin = 0.7978845608028654f * (x + 0.044715f * x * x * x);
    float g = 0.5f * x * (1.f + tanhf(tin));
    G[i] = f2bf(g);
}

// ---------------- host orchestration ----------------
extern "C" void kernel_launch(void* const* d_in, const int* in_sizes, int n_in,
                              void* d_out, int out_size, void* d_ws, size_t ws_size,
                              hipStream_t stream) {
    (void)in_sizes; (void)n_in; (void)out_size; (void)ws_size;

    // types: 0=user, 1=job, 2=skill
    const float* Xin[3] = {(const float*)d_in[0], (const float*)d_in[1], (const float*)d_in[2]};
    const int*   EI[4]  = {(const int*)d_in[3], (const int*)d_in[4],
                           (const int*)d_in[5], (const int*)d_in[6]};
    const int Ncnt[3] = {100000, 50000, 20000};
    const int Din[3]  = {64, 128, 32};
    const int tS[3]   = {2, 0, 1};          // sorted position: job,skill,user
    // edge types in ET order: applies(u->j), rev_applies(j->u), requires(j->s), rev_requires(s->j)
    const int eSrc[4] = {0, 1, 1, 2};
    const int eDst[4] = {1, 0, 2, 1};
    const int ePid[4] = {3, 1, 0, 2};       // sorted key position of each edge type

    // params flatten (JAX sorted-dict order): layers[0], layers[1], lin_in, lin_out
    // per layer (39 leaves): A[4], M[4], Wa[3], Wk[3], Wq[3], Wv[3],
    //                        ba[3], bk[3], bq[3], bv[3], p[4], skip[3]
    const int LIB = 7 + 39 * 2;             // lin_in base = 85
    const float* WinP[3]; const float* binP[3];
    WinP[1] = (const float*)d_in[LIB + 0]; binP[1] = (const float*)d_in[LIB + 1]; // job
    WinP[2] = (const float*)d_in[LIB + 2]; binP[2] = (const float*)d_in[LIB + 3]; // skill
    WinP[0] = (const float*)d_in[LIB + 4]; binP[0] = (const float*)d_in[LIB + 5]; // user
    const float* Wout = (const float*)d_in[LIB + 6];
    const float* bout = (const float*)d_in[LIB + 7];

    // ---- workspace bump allocator ----
    char* wp = (char*)d_ws;
    auto alloc = [&](size_t bytes) -> void* {
        void* p = (void*)wp;
        wp += (bytes + 255) & ~(size_t)255;
        return p;
    };
    unsigned short *Xb[3], *WtIn[3], *Hb[3], *Gb[3];
    unsigned short *WtK[2][3], *WtQ[2][3], *WtV[2][3], *WtA[2][3], *WtO;
    unsigned short *Kb[3], *Qb[3], *Vb[3];
    float *Hf[2][3], *Agg[3], *Ssum[3], *Sc[4];
    unsigned *Mmax[3];
    for (int t = 0; t < 3; ++t) Xb[t]   = (unsigned short*)alloc((size_t)Ncnt[t] * Din[t] * 2);
    for (int t = 0; t < 3; ++t) WtIn[t] = (unsigned short*)alloc((size_t)HIDDIM * Din[t] * 2);
    for (int l = 0; l < 2; ++l)
        for (int t = 0; t < 3; ++t) {
            WtK[l][t] = (unsigned short*)alloc((size_t)HIDDIM * HIDDIM * 2);
            WtQ[l][t] = (unsigned short*)alloc((size_t)HIDDIM * HIDDIM * 2);
            WtV[l][t] = (unsigned short*)alloc((size_t)HIDDIM * HIDDIM * 2);
            WtA[l][t] = (unsigned short*)alloc((size_t)HIDDIM * HIDDIM * 2);
        }
    WtO = (unsigned short*)alloc((size_t)64 * HIDDIM * 2);
    for (int b = 0; b < 2; ++b)
        for (int t = 0; t < 3; ++t) Hf[b][t] = (float*)alloc((size_t)Ncnt[t] * HIDDIM * 4);
    for (int t = 0; t < 3; ++t) Hb[t]  = (unsigned short*)alloc((size_t)Ncnt[t] * HIDDIM * 2);
    for (int t = 0; t < 3; ++t) Gb[t]  = (unsigned short*)alloc((size_t)Ncnt[t] * HIDDIM * 2);
    for (int t = 0; t < 3; ++t) Kb[t]  = (unsigned short*)alloc((size_t)Ncnt[t] * HIDDIM * 2);
    for (int t = 0; t < 3; ++t) Qb[t]  = (unsigned short*)alloc((size_t)Ncnt[t] * HIDDIM * 2);
    for (int t = 0; t < 3; ++t) Vb[t]  = (unsigned short*)alloc((size_t)Ncnt[t] * HIDDIM * 2);
    for (int t = 0; t < 3; ++t) Agg[t]  = (float*)alloc((size_t)Ncnt[t] * HIDDIM * 4);
    for (int t = 0; t < 3; ++t) Ssum[t] = (float*)alloc((size_t)Ncnt[t] * NHEAD * 4);
    for (int t = 0; t < 3; ++t) Mmax[t] = (unsigned*)alloc((size_t)Ncnt[t] * NHEAD * 4);
    for (int et = 0; et < 4; ++et) Sc[et] = (float*)alloc((size_t)NEDGE * NHEAD * 4);

    #define G256(n) dim3(((n) + 255) / 256), dim3(256), 0, stream
    // RT=2 row tiles per wave, 8 waves: 256 rows per block
    #define GGEMM(M) dim3(((M) + 255) / 256), dim3(256), 0, stream
    const size_t QKV_LDS = (size_t)3 * HIDDIM * HIDDIM * 2;   // 96KB dynamic LDS

    // ---- convert inputs & weights to bf16 (weights transposed) ----
    for (int t = 0; t < 3; ++t) {
        int n = Ncnt[t] * Din[t];
        k_cvt<<<G256(n)>>>(Xin[t], Xb[t], n);
        k_wcvt<<<G256(Din[t] * HIDDIM)>>>(WinP[t], WtIn[t], Din[t], HIDDIM);
    }
    k_wcvt<<<G256(HIDDIM * 64)>>>(Wout, WtO, HIDDIM, 64);
    for (int l = 0; l < 2; ++l) {
        int lb = 7 + 39 * l;
        for (int t = 0; t < 3; ++t) {
            k_wcvt<<<G256(HIDDIM * HIDDIM)>>>((const float*)d_in[lb + 11 + tS[t]], WtK[l][t], HIDDIM, HIDDIM);
            k_wcvt<<<G256(HIDDIM * HIDDIM)>>>((const float*)d_in[lb + 14 + tS[t]], WtQ[l][t], HIDDIM, HIDDIM);
            k_wcvt<<<G256(HIDDIM * HIDDIM)>>>((const float*)d_in[lb + 17 + tS[t]], WtV[l][t], HIDDIM, HIDDIM);
            k_wcvt<<<G256(HIDDIM * HIDDIM)>>>((const float*)d_in[lb +  8 + tS[t]], WtA[l][t], HIDDIM, HIDDIM);
        }
    }

    // ---- lin_in: H0 = relu(x @ Win + bin) ----
    gemm_wmma< 64, 128, 2, true, false><<<GGEMM(Ncnt[0])>>>(Xb[0], WtIn[0], binP[0], nullptr, nullptr, Hf[0][0], Hb[0], Ncnt[0]);
    gemm_wmma<128, 128, 2, true, false><<<GGEMM(Ncnt[1])>>>(Xb[1], WtIn[1], binP[1], nullptr, nullptr, Hf[0][1], Hb[1], Ncnt[1]);
    gemm_wmma< 32, 128, 2, true, false><<<GGEMM(Ncnt[2])>>>(Xb[2], WtIn[2], binP[2], nullptr, nullptr, Hf[0][2], Hb[2], Ncnt[2]);

    // ---- HGT layers ----
    int cur = 0;
    for (int l = 0; l < 2; ++l) {
        int lb = 7 + 39 * l;
        // fused K/Q/V projection, bf16 outputs
        for (int t = 0; t < 3; ++t)
            gemm_qkv<128, 128, 2><<<dim3((Ncnt[t] + 255) / 256), dim3(256), QKV_LDS, stream>>>(
                Hb[t], WtK[l][t], WtQ[l][t], WtV[l][t],
                (const float*)d_in[lb + 23 + tS[t]],
                (const float*)d_in[lb + 26 + tS[t]],
                (const float*)d_in[lb + 29 + tS[t]],
                Kb[t], Qb[t], Vb[t], Ncnt[t]);
        for (int t = 0; t < 3; ++t) {
            k_fill_u32<<<G256(Ncnt[t] * NHEAD)>>>(Mmax[t], 0x007FFFFFu, Ncnt[t] * NHEAD); // ord(-inf)
            k_fill_f32<<<G256(Ncnt[t] * NHEAD)>>>(Ssum[t], 0.f, Ncnt[t] * NHEAD);
            k_fill_f32<<<G256(Ncnt[t] * HIDDIM)>>>(Agg[t], 0.f, Ncnt[t] * HIDDIM);
        }
        for (int et = 0; et < 4; ++et)
            k_edgeA<<<G256(NEDGE * NHEAD)>>>(EI[et], Kb[eSrc[et]], Qb[eDst[et]],
                (const float*)d_in[lb + 0 + ePid[et]],
                (const float*)d_in[lb + 32 + ePid[et]],
                Sc[et], Mmax[eDst[et]], NEDGE);
        for (int et = 0; et < 4; ++et)
            k_edgeB<<<G256(NEDGE * NHEAD)>>>(EI[et], Vb[eSrc[et]],
                (const float*)d_in[lb + 4 + ePid[et]],
                Sc[et], Mmax[eDst[et]], Ssum[eDst[et]], Agg[eDst[et]], NEDGE);
        for (int t = 0; t < 3; ++t)
            k_gelu<<<G256(Ncnt[t] * HIDDIM)>>>(Agg[t], Ssum[t], Gb[t], Ncnt[t]);
        for (int t = 0; t < 3; ++t)
            gemm_wmma<128, 128, 2, true, true><<<GGEMM(Ncnt[t])>>>(
                Gb[t], WtA[l][t], (const float*)d_in[lb + 20 + tS[t]],
                Hf[cur][t], (const float*)d_in[lb + 36 + tS[t]],
                Hf[1 - cur][t], Hb[t], Ncnt[t]);
        cur ^= 1;
    }

    // ---- lin_out into d_out (user, job, skill concatenated) ----
    float* out = (float*)d_out;
    gemm_wmma<128, 64, 2, false, false><<<GGEMM(Ncnt[0])>>>(Hb[0], WtO, bout, nullptr, nullptr, out, nullptr, Ncnt[0]);
    gemm_wmma<128, 64, 2, false, false><<<GGEMM(Ncnt[1])>>>(Hb[1], WtO, bout, nullptr, nullptr, out + (size_t)100000 * 64, nullptr, Ncnt[1]);
    gemm_wmma<128, 64, 2, false, false><<<GGEMM(Ncnt[2])>>>(Hb[2], WtO, bout, nullptr, nullptr, out + (size_t)150000 * 64, nullptr, Ncnt[2]);

    #undef G256
    #undef GGEMM
}
